// AutoregressivePredictor_61418032333445
// MI455X (gfx1250) — matrix operand order repaired and checked by reading, not compiled, not run on hardware
//
#include <hip/hip_runtime.h>
#include <cstdint>
#include <cstddef>

// ---------------------------------------------------------------------------
// AutoregressivePredictor for MI455X (gfx1250, wave32, WMMA)
// KV-cache decoding: per step only the new position is computed.
// GEMMs: bf16 A/B, f32 accum via v_wmma_f32_16x16x32_bf16, 128x128 block
// tiles, double-buffered LDS, register-prefetch pipelined, global_prefetch_b8
// deep prefetch into cache.
// ---------------------------------------------------------------------------

typedef __attribute__((ext_vector_type(16))) __bf16 v16bf;
typedef __attribute__((ext_vector_type(8)))  __bf16 v8bf;
typedef __attribute__((ext_vector_type(8)))  float  v8f;
typedef __attribute__((ext_vector_type(4)))  int    v4i;

#define T_TOK   4096
#define HDIM    1024
#define NHEAD   16
#define HEADD   64
#define IDIM    4096
#define VDIM    1024
#define NSTEP   8

// bf16 <-> f32 helpers (storage type = unsigned short, RNE rounding)
__device__ __forceinline__ unsigned short f2bf(float f) {
  unsigned u = __builtin_bit_cast(unsigned, f);
  u += 0x7fffu + ((u >> 16) & 1u);
  return (unsigned short)(u >> 16);
}
__device__ __forceinline__ float bf2f(unsigned short h) {
  unsigned u = ((unsigned)h) << 16;
  return __builtin_bit_cast(float, u);
}

// ---------------------------------------------------------------------------
// WMMA GEMM:  C[M,N] (+)= A[M,K]_bf16(row-major) x B[K,N]_bf16(row-major)
// Block tile 128x128, 8 waves (2x4), each wave 64x32 = 4x2 WMMA tiles.
// A-fragment: lane L -> row (L&15), K base (L>>4)*8, two 16B LDS reads.
// B-fragment: lane L -> N col (L&15), K base (L>>4)*8, +0..7 and +16..23 (same k order as A).
// Double-buffered LDS: one barrier per K-step; ds_store of tile k+1 overlaps
// the WMMAs of tile k.
// ---------------------------------------------------------------------------
#define LDS_A_STRIDE 40   // 32 + 8 pad elements (80B rows, 16B aligned, no bank conflicts)
#define LDS_B_STRIDE 136  // 128 + 8 pad elements (272B rows, 16B aligned)
#define PF_DIST 4         // deep-prefetch distance in K-tiles

template<bool RESID, bool WF32, bool WBF16, bool BIAS>
__global__ __launch_bounds__(256) void gemm_bf16_wmma(
    const unsigned short* __restrict__ A,
    const unsigned short* __restrict__ B,
    float* __restrict__ Cf, unsigned short* __restrict__ Cb,
    const float* __restrict__ bias,
    int M, int N, int K)
{
  __shared__ __align__(16) unsigned short sA[2][128 * LDS_A_STRIDE];
  __shared__ __align__(16) unsigned short sB[2][32 * LDS_B_STRIDE];

  const int tid  = threadIdx.x;
  const int lane = tid & 31;
  const int wave = tid >> 5;
  const int wm   = (wave >> 2) * 64;   // wave M offset in block
  const int wn   = (wave & 3) * 32;    // wave N offset in block
  const int bM   = blockIdx.y * 128;
  const int bN   = blockIdx.x * 128;

  // Global->LDS staging coords
  const int arow = tid >> 1;           // 0..127
  const int acol = (tid & 1) * 16;     // element offset in 32-wide K tile
  const unsigned short* gA = A + (size_t)(bM + arow) * K + acol;

  const int brow = tid >> 4;           // 0..15 (second chunk at +16)
  const int bcol = (tid & 15) * 8;     // element offset in 128-wide N tile
  const unsigned short* gB = B + (size_t)brow * N + bN + bcol;

  const int nk = K / 32;

  // load K-tile 0 into registers and stage into LDS buffer 0
  v4i ra0 = *(const v4i*)(gA);
  v4i ra1 = *(const v4i*)(gA + 8);
  v4i rb0 = *(const v4i*)(gB);
  v4i rb1 = *(const v4i*)(gB + (size_t)16 * N);
  *(v4i*)(sA[0] + arow * LDS_A_STRIDE + acol)        = ra0;
  *(v4i*)(sA[0] + arow * LDS_A_STRIDE + acol + 8)    = ra1;
  *(v4i*)(sB[0] + brow * LDS_B_STRIDE + bcol)        = rb0;
  *(v4i*)(sB[0] + (brow + 16) * LDS_B_STRIDE + bcol) = rb1;
  __syncthreads();

  v8f acc[4][2] = {};

  for (int kb = 0; kb < nk; ++kb) {
    const unsigned short* curA = sA[kb & 1];
    const unsigned short* curB = sB[kb & 1];
    const bool more = (kb + 1 < nk);

    // register prefetch of next K-tile (latency hidden under WMMAs)
    if (more) {
      const unsigned short* pA = gA + (size_t)(kb + 1) * 32;
      ra0 = *(const v4i*)(pA);
      ra1 = *(const v4i*)(pA + 8);
      const unsigned short* pB = gB + (size_t)(kb + 1) * 32 * N;
      rb0 = *(const v4i*)(pB);
      rb1 = *(const v4i*)(pB + (size_t)16 * N);
    }
    // deep prefetch (global_prefetch_b8) a few K-tiles ahead
    if (kb + PF_DIST < nk) {
      __builtin_prefetch(gA + (size_t)(kb + PF_DIST) * 32, 0, 1);
      __builtin_prefetch(gB + (size_t)(kb + PF_DIST) * 32 * N, 0, 1);
    }

    // A fragments: 4 x 16x32
    v16bf af[4];
#pragma unroll
    for (int i = 0; i < 4; ++i) {
      const unsigned short* p =
          curA + (wm + i * 16 + (lane & 15)) * LDS_A_STRIDE + ((lane >> 4) * 8);
      v8bf lo = *(const v8bf*)p;        // K base..base+7   -> VGPR0-3
      v8bf hi = *(const v8bf*)(p + 16); // K base+16..+23   -> VGPR4-7
      af[i] = __builtin_shufflevector(lo, hi, 0, 1, 2, 3, 4, 5, 6, 7,
                                      8, 9, 10, 11, 12, 13, 14, 15);
    }
    // B fragments: 2 x 32x16 (lane = N column, 16 K values strided in LDS)
    v16bf bfr[2];
#pragma unroll
    for (int j = 0; j < 2; ++j) {
      const unsigned short* p = curB + ((lane >> 4) * 8) * LDS_B_STRIDE + wn + j * 16 + (lane & 15);
#pragma unroll
      for (int s = 0; s < 16; ++s)
        bfr[j][s] = __builtin_bit_cast(__bf16, p[((s & 7) + (s >> 3) * 16) * LDS_B_STRIDE]);
    }

#pragma unroll
    for (int i = 0; i < 4; ++i)
#pragma unroll
      for (int j = 0; j < 2; ++j)
        acc[i][j] = __builtin_amdgcn_wmma_f32_16x16x32_bf16(
            false, af[i], false, bfr[j], (short)0, acc[i][j], false, false);

    // stage next K-tile into the other LDS buffer, then single barrier
    if (more) {
      unsigned short* nA = sA[(kb + 1) & 1];
      unsigned short* nB = sB[(kb + 1) & 1];
      *(v4i*)(nA + arow * LDS_A_STRIDE + acol)        = ra0;
      *(v4i*)(nA + arow * LDS_A_STRIDE + acol + 8)    = ra1;
      *(v4i*)(nB + brow * LDS_B_STRIDE + bcol)        = rb0;
      *(v4i*)(nB + (brow + 16) * LDS_B_STRIDE + bcol) = rb1;
      __syncthreads();
    }
  }

  // Epilogue per C/D layout: VGPR r, lanes 0-15 -> M=r N=lane; lanes 16-31 -> M=r+8
  const int rhalf = (lane >> 4) * 8;
  const int cc    = lane & 15;
#pragma unroll
  for (int i = 0; i < 4; ++i) {
#pragma unroll
    for (int j = 0; j < 2; ++j) {
      const int gn = bN + wn + j * 16 + cc;
#pragma unroll
      for (int r = 0; r < 8; ++r) {
        const int gm = bM + wm + i * 16 + rhalf + r;
        float v = acc[i][j][r];
        if constexpr (BIAS) v += bias[gn];
        const size_t off = (size_t)gm * N + gn;
        if constexpr (RESID) v += Cf[off];
        if constexpr (WF32)  Cf[off] = v;
        if constexpr (WBF16) Cb[off] = f2bf(v);
      }
    }
  }
}

// ---------------------------------------------------------------------------
// fp32 -> bf16 weight convert (with column-concat for packed QKV / gate-up)
// ---------------------------------------------------------------------------
__global__ __launch_bounds__(256) void convert_cat_k(
    const float* __restrict__ W, unsigned short* __restrict__ dst,
    int K, int N, int ld, int colOff)
{
  size_t idx = (size_t)blockIdx.x * 256 + threadIdx.x;
  if (idx >= (size_t)K * N) return;
  int k = (int)(idx / N), n = (int)(idx % N);
  dst[(size_t)k * ld + colOff + n] = f2bf(W[idx]);
}

__global__ __launch_bounds__(256) void copy_f32_k(
    const float* __restrict__ src, float* __restrict__ dst, int n)
{
  int idx = blockIdx.x * 256 + threadIdx.x;
  if (idx < n) dst[idx] = src[idx];
}

// ---------------------------------------------------------------------------
// RMSNorm: one block per row (H=1024), wave32 shuffle + LDS cross-wave reduce
// ---------------------------------------------------------------------------
__global__ __launch_bounds__(256) void rmsnorm_k(
    const float* __restrict__ h, const float* __restrict__ w,
    unsigned short* __restrict__ out)
{
  const int t = blockIdx.x, tid = threadIdx.x;
  const float* x = h + (size_t)t * HDIM;
  float s = 0.f;
#pragma unroll
  for (int j = 0; j < 4; ++j) { float v = x[tid + j * 256]; s += v * v; }
#pragma unroll
  for (int off = 16; off > 0; off >>= 1) s += __shfl_xor(s, off, 32);
  __shared__ float red[8];
  if ((tid & 31) == 0) red[tid >> 5] = s;
  __syncthreads();
  float tot = red[0] + red[1] + red[2] + red[3] +
              red[4] + red[5] + red[6] + red[7];
  float inv = rsqrtf(tot * (1.0f / HDIM) + 1e-6f);
#pragma unroll
  for (int j = 0; j < 4; ++j) {
    int c = tid + j * 256;
    out[(size_t)t * HDIM + c] = f2bf(x[c] * inv * w[c]);
  }
}

// ---------------------------------------------------------------------------
// RoPE(pos=p) on q,k of the new position + write K/V caches (bf16)
// ---------------------------------------------------------------------------
__global__ __launch_bounds__(256) void rope_cache_k(
    const float* __restrict__ qkv, unsigned short* __restrict__ q,
    unsigned short* __restrict__ kc, unsigned short* __restrict__ vc, int p)
{
  const int t = blockIdx.x, tid = threadIdx.x;
  const float* row = qkv + (size_t)t * (3 * HDIM);
  const size_t kvoff = ((size_t)t * NSTEP + p) * HDIM;
#pragma unroll
  for (int s = 0; s < 2; ++s) {
    int pi = tid * 2 + s;            // 512 rotation pairs (16 heads x 32)
    int head = pi >> 5, i = pi & 31;
    float ang = (float)p * __powf(10000.0f, -(2.0f * i) / 64.0f);
    float sn, c;
    __sincosf(ang, &sn, &c);
    int b = head * HEADD + i;
    float q1 = row[b], q2 = row[b + 32];
    q[(size_t)t * HDIM + b]      = f2bf(q1 * c - q2 * sn);
    q[(size_t)t * HDIM + b + 32] = f2bf(q2 * c + q1 * sn);
    float k1 = row[HDIM + b], k2 = row[HDIM + b + 32];
    kc[kvoff + b]      = f2bf(k1 * c - k2 * sn);
    kc[kvoff + b + 32] = f2bf(k2 * c + k1 * sn);
  }
#pragma unroll
  for (int j = 0; j < 4; ++j) {
    int c2 = tid + j * 256;
    vc[kvoff + c2] = f2bf(row[2 * HDIM + c2]);
  }
}

// ---------------------------------------------------------------------------
// Attention: one wave32 per (token, head); <=8 cached keys, lanes over dims
// ---------------------------------------------------------------------------
__global__ __launch_bounds__(256) void attn_k(
    const unsigned short* __restrict__ q, const unsigned short* __restrict__ kc,
    const unsigned short* __restrict__ vc, unsigned short* __restrict__ out, int p)
{
  const int lane = threadIdx.x & 31, wv = threadIdx.x >> 5;
  const int task = blockIdx.x * 8 + wv;          // T*NH tasks
  const int t = task >> 4, head = task & 15;
  const size_t qoff = (size_t)t * HDIM + head * HEADD;
  const float q0 = bf2f(q[qoff + 2 * lane]);
  const float q1 = bf2f(q[qoff + 2 * lane + 1]);
  const int L = p + 1;
  float s[NSTEP];
#pragma unroll
  for (int m = 0; m < NSTEP; ++m) {
    if (m < L) {
      size_t ko = ((size_t)t * NSTEP + m) * HDIM + head * HEADD;
      float d = q0 * bf2f(kc[ko + 2 * lane]) + q1 * bf2f(kc[ko + 2 * lane + 1]);
#pragma unroll
      for (int off = 16; off > 0; off >>= 1) d += __shfl_xor(d, off, 32);
      s[m] = d * 0.125f;               // 1/sqrt(64)
    } else {
      s[m] = -1e30f;
    }
  }
  float mx = s[0];
#pragma unroll
  for (int m = 1; m < NSTEP; ++m) mx = fmaxf(mx, s[m]);
  float sum = 0.f;
#pragma unroll
  for (int m = 0; m < NSTEP; ++m) { s[m] = __expf(s[m] - mx); sum += s[m]; }
  const float inv = 1.0f / sum;
  float c0 = 0.f, c1 = 0.f;
#pragma unroll
  for (int m = 0; m < NSTEP; ++m) {
    if (m < L) {
      size_t vo = ((size_t)t * NSTEP + m) * HDIM + head * HEADD;
      c0 += s[m] * bf2f(vc[vo + 2 * lane]);
      c1 += s[m] * bf2f(vc[vo + 2 * lane + 1]);
    }
  }
  out[qoff + 2 * lane]     = f2bf(c0 * inv);
  out[qoff + 2 * lane + 1] = f2bf(c1 * inv);
}

// ---------------------------------------------------------------------------
// act = silu(gate) * up, from packed (T, 2I) bf16 buffer
// ---------------------------------------------------------------------------
__global__ __launch_bounds__(256) void silu_mul_k(
    const unsigned short* __restrict__ gu, unsigned short* __restrict__ act)
{
  size_t gid = (size_t)blockIdx.x * 256 + threadIdx.x;   // T*I total
  size_t t = gid >> 12;
  size_t i = gid & 4095;
  float g = bf2f(gu[t * (2 * IDIM) + i]);
  float u = bf2f(gu[t * (2 * IDIM) + IDIM + i]);
  float sg = g / (1.0f + __expf(-g));
  act[gid] = f2bf(sg * u);
}

// ---------------------------------------------------------------------------
// argmax over V=1024 logits (first-max tie-break) + next-token embedding
// ---------------------------------------------------------------------------
__global__ __launch_bounds__(256) void argmax_embed_k(
    const float* __restrict__ logits, const float* __restrict__ E,
    int* __restrict__ toks, float* __restrict__ hcur, int p)
{
  const int t = blockIdx.x, tid = threadIdx.x;
  const float* lr = logits + (size_t)t * VDIM;
  float bv = -3.4e38f; int bi = 0;
#pragma unroll
  for (int j = 0; j < 4; ++j) {
    int c = j * 256 + tid;           // ascending scan -> keep first max
    float v = lr[c];
    if (v > bv) { bv = v; bi = c; }
  }
#pragma unroll
  for (int off = 16; off > 0; off >>= 1) {
    float ov = __shfl_xor(bv, off, 32);
    int   oi = __shfl_xor(bi, off, 32);
    if (ov > bv || (ov == bv && oi < bi)) { bv = ov; bi = oi; }
  }
  __shared__ float sv[8];
  __shared__ int   si[8];
  __shared__ int   stok;
  if ((tid & 31) == 0) { sv[tid >> 5] = bv; si[tid >> 5] = bi; }
  __syncthreads();
  if (tid == 0) {
    float v = sv[0]; int ix = si[0];
#pragma unroll
    for (int w = 1; w < 8; ++w)
      if (sv[w] > v || (sv[w] == v && si[w] < ix)) { v = sv[w]; ix = si[w]; }
    stok = ix;
    toks[(size_t)t * NSTEP + p] = ix;
  }
  __syncthreads();
  if (p < NSTEP - 1) {
    const int tok = stok;
#pragma unroll
    for (int j = 0; j < 4; ++j) {
      int c = tid + j * 256;
      hcur[(size_t)t * HDIM + c] = E[(size_t)tok * HDIM + c];
    }
  }
}

// ---------------------------------------------------------------------------
// Host launcher
// ---------------------------------------------------------------------------
extern "C" void kernel_launch(void* const* d_in, const int* in_sizes, int n_in,
                              void* d_out, int out_size, void* d_ws, size_t ws_size,
                              hipStream_t stream) {
  (void)in_sizes; (void)n_in; (void)out_size;
  const float* chunk = (const float*)d_in[0];
  const float* Wq   = (const float*)d_in[1];
  const float* Wk   = (const float*)d_in[2];
  const float* Wv   = (const float*)d_in[3];
  const float* Wo   = (const float*)d_in[4];
  const float* Wg   = (const float*)d_in[5];
  const float* Wu   = (const float*)d_in[6];
  const float* Wd   = (const float*)d_in[7];
  const float* n1   = (const float*)d_in[8];
  const float* n2   = (const float*)d_in[9];
  const float* E    = (const float*)d_in[10];
  const float* Wout = (const float*)d_in[11];
  const float* bout = (const float*)d_in[12];

  char* ws = (char*)d_ws;
  size_t off = 0;
  auto carve = [&](size_t bytes) -> char* {
    char* p = ws + off;
    off += (bytes + 255) & ~(size_t)255;
    return p;
  };
  unsigned short* wqkv = (unsigned short*)carve((size_t)HDIM * 3 * HDIM * 2);   // 6 MB
  unsigned short* wo   = (unsigned short*)carve((size_t)HDIM * HDIM * 2);       // 2 MB
  unsigned short* wgu  = (unsigned short*)carve((size_t)HDIM * 2 * IDIM * 2);   // 16 MB
  unsigned short* wd   = (unsigned short*)carve((size_t)IDIM * HDIM * 2);       // 8 MB
  unsigned short* wout = (unsigned short*)carve((size_t)HDIM * VDIM * 2);       // 2 MB
  float*          hcur = (float*)carve((size_t)T_TOK * HDIM * 4);               // 16 MB
  unsigned short* xnrm = (unsigned short*)carve((size_t)T_TOK * HDIM * 2);      // 8 MB
  unsigned short* atto = (unsigned short*)carve((size_t)T_TOK * HDIM * 2);      // 8 MB
  unsigned short* qbuf = (unsigned short*)carve((size_t)T_TOK * HDIM * 2);      // 8 MB
  unsigned short* kcch = (unsigned short*)carve((size_t)T_TOK * NSTEP * HDIM * 2); // 64 MB
  unsigned short* vcch = (unsigned short*)carve((size_t)T_TOK * NSTEP * HDIM * 2); // 64 MB
  unsigned short* act  = (unsigned short*)carve((size_t)T_TOK * IDIM * 2);      // 32 MB
  char*           big  = carve((size_t)T_TOK * 2 * IDIM * 2);                   // 64 MB
  if (off > ws_size) return;  // workspace too small: bail safely

  float*          qkvf   = (float*)big;           // (T, 3072) f32
  unsigned short* gub    = (unsigned short*)big;  // (T, 8192) bf16
  float*          logits = (float*)big;           // (T, 1024) f32
  int*            toks   = (int*)d_out;           // (T, 8) int32

  const int TB = 256;
  auto g1 = [](size_t n) { return dim3((unsigned)((n + 255) / 256)); };

  // weight converts (fp32 -> bf16, column-packed where useful)
  convert_cat_k<<<g1((size_t)HDIM * HDIM), TB, 0, stream>>>(Wq, wqkv, HDIM, HDIM, 3 * HDIM, 0);
  convert_cat_k<<<g1((size_t)HDIM * HDIM), TB, 0, stream>>>(Wk, wqkv, HDIM, HDIM, 3 * HDIM, HDIM);
  convert_cat_k<<<g1((size_t)HDIM * HDIM), TB, 0, stream>>>(Wv, wqkv, HDIM, HDIM, 3 * HDIM, 2 * HDIM);
  convert_cat_k<<<g1((size_t)HDIM * HDIM), TB, 0, stream>>>(Wo, wo, HDIM, HDIM, HDIM, 0);
  convert_cat_k<<<g1((size_t)HDIM * IDIM), TB, 0, stream>>>(Wg, wgu, HDIM, IDIM, 2 * IDIM, 0);
  convert_cat_k<<<g1((size_t)HDIM * IDIM), TB, 0, stream>>>(Wu, wgu, HDIM, IDIM, 2 * IDIM, IDIM);
  convert_cat_k<<<g1((size_t)IDIM * HDIM), TB, 0, stream>>>(Wd, wd, IDIM, HDIM, HDIM, 0);
  convert_cat_k<<<g1((size_t)HDIM * VDIM), TB, 0, stream>>>(Wout, wout, HDIM, VDIM, VDIM, 0);
  copy_f32_k<<<g1((size_t)T_TOK * HDIM), TB, 0, stream>>>(chunk, hcur, T_TOK * HDIM);

  for (int p = 0; p < NSTEP; ++p) {
    rmsnorm_k<<<T_TOK, TB, 0, stream>>>(hcur, n1, xnrm);
    gemm_bf16_wmma<false, true, false, false>
        <<<dim3(3 * HDIM / 128, T_TOK / 128), TB, 0, stream>>>(
            xnrm, wqkv, qkvf, nullptr, nullptr, T_TOK, 3 * HDIM, HDIM);
    rope_cache_k<<<T_TOK, TB, 0, stream>>>(qkvf, qbuf, kcch, vcch, p);
    attn_k<<<T_TOK * NHEAD / 8, TB, 0, stream>>>(qbuf, kcch, vcch, atto, p);
    gemm_bf16_wmma<true, true, false, false>
        <<<dim3(HDIM / 128, T_TOK / 128), TB, 0, stream>>>(
            atto, wo, hcur, nullptr, nullptr, T_TOK, HDIM, HDIM);
    rmsnorm_k<<<T_TOK, TB, 0, stream>>>(hcur, n2, xnrm);
    gemm_bf16_wmma<false, false, true, false>
        <<<dim3(2 * IDIM / 128, T_TOK / 128), TB, 0, stream>>>(
            xnrm, wgu, nullptr, gub, nullptr, T_TOK, 2 * IDIM, HDIM);
    silu_mul_k<<<g1((size_t)T_TOK * IDIM), TB, 0, stream>>>(gub, act);
    // down-proj: residual into hcur (f32) and bf16 copy into xnrm for logits
    gemm_bf16_wmma<true, true, true, false>
        <<<dim3(HDIM / 128, T_TOK / 128), TB, 0, stream>>>(
            act, wd, hcur, xnrm, nullptr, T_TOK, HDIM, IDIM);
    gemm_bf16_wmma<false, true, false, true>
        <<<dim3(VDIM / 128, T_TOK / 128), TB, 0, stream>>>(
            xnrm, wout, logits, nullptr, bout, T_TOK, VDIM, HDIM);
    argmax_embed_k<<<T_TOK, TB, 0, stream>>>(logits, E, toks, hcur, p);
  }
}
